// MultiHeadAttention_31035433681291
// MI455X (gfx1250) — matrix operand order repaired and checked by reading, not compiled, not run on hardware
//
#include <hip/hip_runtime.h>

// ---------------------------------------------------------------------------
// Multi-head attention forward for gfx1250 (MI455X), f16 WMMA with f32 accum.
// d_out = [ out: 2*2048*1024 f32 | attn: 2*16*2048*2048 f32 ]
// d_ws  = 48 MB of f16 scratch (xh, Qh, Kh, Vth, ctxh, Wq/Wk/Wv/Wo in f16).
//
// Fused attention core: one workgroup owns a 32-row q-block, holds the whole
// 32x2048 f32 score panel in LDS (257 KB -- CDNA5's 320KB/WGP LDS), so the
// attn matrix touches HBM exactly once (the mandatory 537 MB f32 output).
// ---------------------------------------------------------------------------

typedef _Float16 f16;
typedef __attribute__((ext_vector_type(16))) _Float16 v16h;
typedef __attribute__((ext_vector_type(8)))  _Float16 v8h;
typedef __attribute__((ext_vector_type(8)))  float    v8f;
typedef __attribute__((ext_vector_type(4)))  float    v4f;

#define WMMA_F16(a, b, c) \
  __builtin_amdgcn_wmma_f32_16x16x32_f16(false, (a), false, (b), (short)0, (c), false, false)

namespace {

constexpr int Bn = 2, S = 2048, D = 1024, H = 16, HD = 64;
constexpr int MROWS = Bn * S;  // 4096

// ------------------------- fragment loaders (wave32) ------------------------
// A: 16x32 f16 tile from row-major [*, lda]. Lane l = row (l&15);
//    VGPR0-3 <- K kb..kb+7, VGPR4-7 <- K kb+16..kb+23, kb = (l>>4)*8.
__device__ __forceinline__ v16h load_a_f16(const f16* A, int lda, int lane) {
  const f16* p = A + (size_t)(lane & 15) * lda + ((lane >> 4) << 3);
  v8h lo = *(const v8h*)(p);
  v8h hi = *(const v8h*)(p + 16);
  v16h r;
#pragma unroll
  for (int i = 0; i < 8; ++i) { r[i] = lo[i]; r[i + 8] = hi[i]; }
  return r;
}

// Same A tile sourced from f32 (LDS score panel), converted on the fly.
__device__ __forceinline__ v16h load_a_f32(const float* A, int lda, int lane) {
  const float* p = A + (size_t)(lane & 15) * lda + ((lane >> 4) << 3);
  v16h r;
#pragma unroll
  for (int i = 0; i < 8; ++i) {
    r[i]     = (f16)p[i];
    r[i + 8] = (f16)p[16 + i];
  }
  return r;
}

// B: 32x16 tile where B = (stored operand)^T, operand row-major [N, ldb].
//    Lane l = column (l&15); VGPR0-3 <- K kb..kb+7, VGPR4-7 <- K kb+16..kb+23, kb = (l>>4)*8.
__device__ __forceinline__ v16h load_b_f16(const f16* Bm, int ldb, int lane) {
  const f16* p = Bm + (size_t)(lane & 15) * ldb + ((lane >> 4) << 3);
  v8h lo = *(const v8h*)(p);
  v8h hi = *(const v8h*)(p + 16);
  v16h r;
#pragma unroll
  for (int i = 0; i < 8; ++i) { r[i] = lo[i]; r[i + 8] = hi[i]; }
  return r;
}

struct Acc { v8f c[2][2]; };

// 32x32 output tile per wave: C = A[32 x K] * Bop[32 x K]^T, f32 accumulate.
__device__ __forceinline__ void gemm_f16(const f16* A, int lda, const f16* Bm,
                                         int ldb, int K, int lane, Acc& acc) {
  for (int k0 = 0; k0 < K; k0 += 32) {
    v16h a0 = load_a_f16(A + k0, lda, lane);
    v16h a1 = load_a_f16(A + (size_t)16 * lda + k0, lda, lane);
    v16h b0 = load_b_f16(Bm + k0, ldb, lane);
    v16h b1 = load_b_f16(Bm + (size_t)16 * ldb + k0, ldb, lane);
    acc.c[0][0] = WMMA_F16(a0, b0, acc.c[0][0]);
    acc.c[0][1] = WMMA_F16(a0, b1, acc.c[0][1]);
    acc.c[1][0] = WMMA_F16(a1, b0, acc.c[1][0]);
    acc.c[1][1] = WMMA_F16(a1, b1, acc.c[1][1]);
  }
}

}  // namespace

// --------------------------------- kernels ----------------------------------

__global__ void cvt_f32_to_f16(const float* __restrict__ src,
                               f16* __restrict__ dst, int n) {
  int i = blockIdx.x * blockDim.x + threadIdx.x;
  if (i < n) dst[i] = (f16)src[i];
}

// Y = Xh @ W^T + bias; output reshaped per-head.
// transposeOut=0: Out[((b*H+h)*S + s)*HD + hd]   (Q, K)
// transposeOut=1: Out[((b*H+h)*HD + hd)*S + s]   (V stored transposed for ctx)
__global__ void proj_kernel(const f16* __restrict__ X, const f16* __restrict__ W,
                            const float* __restrict__ bias, f16* __restrict__ Out,
                            int transposeOut) {
  int wave = blockIdx.x * (blockDim.x >> 5) + (threadIdx.x >> 5);
  int lane = threadIdx.x & 31;
  const int tilesN = D / 32;  // 32
  int M0 = (wave / tilesN) * 32;
  int N0 = (wave % tilesN) * 32;
  Acc acc = {};
  gemm_f16(X + (size_t)M0 * D, D, W + (size_t)N0 * D, D, D, lane, acc);
  int colq = lane & 15, rbase = (lane >> 4) * 8;
#pragma unroll
  for (int i = 0; i < 2; ++i)
#pragma unroll
    for (int j = 0; j < 2; ++j)
#pragma unroll
      for (int r = 0; r < 8; ++r) {
        int m = M0 + i * 16 + rbase + r;
        int e = N0 + j * 16 + colq;
        float v = acc.c[i][j][r] + bias[e];
        int b = m >> 11, s = m & (S - 1);
        int h = e >> 6, hd = e & (HD - 1);
        size_t idx = transposeOut
                         ? ((size_t)((b * H + h) * HD + hd)) * S + s
                         : ((size_t)((b * H + h) * S + s)) * HD + hd;
        Out[idx] = (f16)v;
      }
}

// Fused scores -> softmax -> attn-write -> ctx for one (bh, 32-row q-block).
// 256 threads = 8 waves. LDS holds the full 32x2048 f32 score panel.
__global__ void attn_fused_kernel(const f16* __restrict__ Qh,
                                  const f16* __restrict__ Kh,
                                  const f16* __restrict__ Vth,
                                  float* __restrict__ attn,
                                  f16* __restrict__ ctxh) {
  constexpr int SL = S + 8;          // padded LDS row stride (floats)
  __shared__ float sS[32 * SL];      // 263168 bytes (CDNA5: 320KB/WG legal)

  const int bh = blockIdx.y;
  const int qt = blockIdx.x;         // q-tile index (32 rows each)
  const int wv = threadIdx.x >> 5;   // wave 0..7
  const int lane = threadIdx.x & 31;

  const f16* Qbase = Qh + (size_t)bh * S * HD + (size_t)qt * 32 * HD;
  const f16* Kbase = Kh + (size_t)bh * S * HD;
  const f16* Vbase = Vth + (size_t)bh * HD * S;
  float* attnBlk = attn + (size_t)bh * S * S + (size_t)qt * 32 * S;

  // ---------------- pass 1: scores = (Q Kt)/8 -> LDS --------------------
  // Q A-fragments for all 32 rows, K = 0..63, held in registers.
  v16h qa00 = load_a_f16(Qbase, HD, lane);
  v16h qa01 = load_a_f16(Qbase + 32, HD, lane);
  v16h qa10 = load_a_f16(Qbase + 16 * HD, HD, lane);
  v16h qa11 = load_a_f16(Qbase + 16 * HD + 32, HD, lane);

  const int coln = lane & 15, rbase = (lane >> 4) * 8;
#pragma unroll 1
  for (int t = 0; t < 8; ++t) {
    int N0 = (wv * 8 + t) * 32;  // this wave's key tile
    const f16* Kp = Kbase + (size_t)N0 * HD;
    v16h b00 = load_b_f16(Kp, HD, lane);
    v16h b01 = load_b_f16(Kp + 32, HD, lane);
    v16h b10 = load_b_f16(Kp + 16 * HD, HD, lane);
    v16h b11 = load_b_f16(Kp + 16 * HD + 32, HD, lane);
    v8f c00 = {}, c01 = {}, c10 = {}, c11 = {};
    c00 = WMMA_F16(qa00, b00, c00); c00 = WMMA_F16(qa01, b01, c00);
    c01 = WMMA_F16(qa00, b10, c01); c01 = WMMA_F16(qa01, b11, c01);
    c10 = WMMA_F16(qa10, b00, c10); c10 = WMMA_F16(qa11, b01, c10);
    c11 = WMMA_F16(qa10, b10, c11); c11 = WMMA_F16(qa11, b11, c11);
#pragma unroll
    for (int r = 0; r < 8; ++r) {
      sS[(size_t)(rbase + r) * SL + N0 + coln]           = c00[r] * 0.125f;
      sS[(size_t)(rbase + r) * SL + N0 + 16 + coln]      = c01[r] * 0.125f;
      sS[(size_t)(16 + rbase + r) * SL + N0 + coln]      = c10[r] * 0.125f;
      sS[(size_t)(16 + rbase + r) * SL + N0 + 16 + coln] = c11[r] * 0.125f;
    }
  }
  __syncthreads();

  // ------------- pass 2: softmax rows (4 rows per wave), write attn -------
#pragma unroll 1
  for (int rr = 0; rr < 4; ++rr) {
    int row = wv * 4 + rr;
    float* p = sS + (size_t)row * SL;
    float m = -3.402823466e38f;
    for (int i = lane; i < S; i += 32) m = fmaxf(m, p[i]);
#pragma unroll
    for (int off = 16; off > 0; off >>= 1) m = fmaxf(m, __shfl_xor(m, off, 32));
    float sum = 0.f;
    for (int i = lane; i < S; i += 32) {
      float e = __expf(p[i] - m);
      p[i] = e;
      sum += e;
    }
#pragma unroll
    for (int off = 16; off > 0; off >>= 1) sum += __shfl_xor(sum, off, 32);
    float inv = 1.0f / sum;
    float* g = attnBlk + (size_t)row * S;
    for (int i = lane; i < S; i += 32) {
      float v = p[i] * inv;
      p[i] = v;   // normalized panel stays in LDS for ctx
      g[i] = v;   // single mandatory HBM write of attn
    }
  }
  __syncthreads();

  // ------ pass 3: ctx = attn @ V, K split 8 ways across the waves ---------
  v8f acc[2][4] = {};
#pragma unroll 1
  for (int kk = 0; kk < 256; kk += 32) {
    int k0 = wv * 256 + kk;
    v16h a0 = load_a_f32(sS + k0, SL, lane);
    v16h a1 = load_a_f32(sS + (size_t)16 * SL + k0, SL, lane);
#pragma unroll
    for (int j = 0; j < 4; ++j) {
      v16h b = load_b_f16(Vbase + (size_t)(j * 16) * S + k0, S, lane);
      acc[0][j] = WMMA_F16(a0, b, acc[0][j]);
      acc[1][j] = WMMA_F16(a1, b, acc[1][j]);
    }
  }
  __syncthreads();  // score panel reads complete; reuse LDS for partials

  // partial C[wv][32][64] -> LDS
#pragma unroll
  for (int i = 0; i < 2; ++i)
#pragma unroll
    for (int j = 0; j < 4; ++j)
#pragma unroll
      for (int r = 0; r < 8; ++r) {
        int m = i * 16 + rbase + r;
        int n = j * 16 + coln;
        sS[(size_t)wv * 2048 + m * 64 + n] = acc[i][j][r];
      }
  __syncthreads();

  // reduce 8 partials, store ctx in f16 [B*S, D] with head interleave
  int b = bh >> 4, h = bh & 15;
  for (int e = threadIdx.x; e < 32 * 64; e += 256) {
    int m = e >> 6, n = e & 63;
    float s = 0.f;
#pragma unroll
    for (int w = 0; w < 8; ++w) s += sS[(size_t)w * 2048 + m * 64 + n];
    int q = qt * 32 + m;
    ctxh[((size_t)(b * S + q)) * D + h * HD + n] = (f16)s;
  }
}

// out = ctx @ Wo^T + bo -> f32 into d_out[0 .. 4194304)
__global__ void outproj_kernel(const f16* __restrict__ ctxh,
                               const f16* __restrict__ Woh,
                               const float* __restrict__ bo,
                               float* __restrict__ out) {
  int wave = blockIdx.x * (blockDim.x >> 5) + (threadIdx.x >> 5);
  int lane = threadIdx.x & 31;
  const int tilesN = D / 32;  // 32
  int M0 = (wave / tilesN) * 32;
  int N0 = (wave % tilesN) * 32;
  Acc acc = {};
  gemm_f16(ctxh + (size_t)M0 * D, D, Woh + (size_t)N0 * D, D, D, lane, acc);
  int colq = lane & 15, rbase = (lane >> 4) * 8;
#pragma unroll
  for (int i = 0; i < 2; ++i)
#pragma unroll
    for (int j = 0; j < 2; ++j)
#pragma unroll
      for (int r = 0; r < 8; ++r) {
        int m = M0 + i * 16 + rbase + r;
        int e = N0 + j * 16 + colq;
        out[(size_t)m * D + e] = acc.c[i][j][r] + bo[e];
      }
}

// --------------------------------- launcher ---------------------------------

extern "C" void kernel_launch(void* const* d_in, const int* in_sizes, int n_in,
                              void* d_out, int out_size, void* d_ws,
                              size_t ws_size, hipStream_t stream) {
  (void)in_sizes; (void)n_in; (void)out_size; (void)ws_size;

  const float* x  = (const float*)d_in[0];
  const float* Wq = (const float*)d_in[1];
  const float* bq = (const float*)d_in[2];
  const float* Wk = (const float*)d_in[3];
  const float* bk = (const float*)d_in[4];
  const float* Wv = (const float*)d_in[5];
  const float* bv = (const float*)d_in[6];
  const float* Wo = (const float*)d_in[7];
  const float* bo = (const float*)d_in[8];

  // f16 scratch layout (element offsets); total 48 MB.
  f16* ws = (f16*)d_ws;
  f16* xh   = ws;                   // [B*S, D]
  f16* Qh   = ws + 4194304;         // [B,H,S,HD]
  f16* Kh   = ws + 8388608;         // [B,H,S,HD]
  f16* Vth  = ws + 12582912;        // [B,H,HD,S]
  f16* ctxh = ws + 16777216;        // [B*S, D]
  f16* Wqh  = ws + 20971520;        // [D,D]
  f16* Wkh  = ws + 22020096;
  f16* Wvh  = ws + 23068672;
  f16* Woh  = ws + 24117248;

  float* out  = (float*)d_out;
  float* attn = out + (size_t)MROWS * D;  // 2*16*2048*2048 f32

  const int NX = MROWS * D;   // 4194304
  const int NW = D * D;       // 1048576

  cvt_f32_to_f16<<<NX / 256, 256, 0, stream>>>(x,  xh,  NX);
  cvt_f32_to_f16<<<NW / 256, 256, 0, stream>>>(Wq, Wqh, NW);
  cvt_f32_to_f16<<<NW / 256, 256, 0, stream>>>(Wk, Wkh, NW);
  cvt_f32_to_f16<<<NW / 256, 256, 0, stream>>>(Wv, Wvh, NW);
  cvt_f32_to_f16<<<NW / 256, 256, 0, stream>>>(Wo, Woh, NW);

  // Projections: 4096x1024x1024 each; 4096 waves -> 1024 blocks of 4 waves.
  proj_kernel<<<1024, 128, 0, stream>>>(xh, Wqh, bq, Qh, 0);
  proj_kernel<<<1024, 128, 0, stream>>>(xh, Wkh, bk, Kh, 0);
  proj_kernel<<<1024, 128, 0, stream>>>(xh, Wvh, bv, Vth, 1);

  // Fused attention: grid = (q-tiles, b*h) = (64, 32), 8 waves per block.
  attn_fused_kernel<<<dim3(S / 32, Bn * H), 256, 0, stream>>>(Qh, Kh, Vth,
                                                              attn, ctxh);

  // Output projection: 4096x1024x1024 + bias -> d_out.
  outproj_kernel<<<1024, 128, 0, stream>>>(ctxh, Woh, bo, out);
}
